// DGLTreeLSTM_66683662237734
// MI455X (gfx1250) — matrix operand order repaired and checked by reading, not compiled, not run on hardware
//
#include <hip/hip_runtime.h>

#define TREES   32
#define NPG_    4095
#define NTOT    131040
#define HS      256
#define ES      256
#define H3      768
#define NCLS    104
#define VOCAB_  20000
#define PADTOK  19999
#define LDA     264   // padded LDS row stride (u16): 528B = 33*16B, conflict-free

typedef __attribute__((ext_vector_type(16))) __bf16 v16bf;
typedef __attribute__((ext_vector_type(16))) short  v16s;
typedef __attribute__((ext_vector_type(8)))  float  v8f;

union FragAB { v16s s; v16bf b; uint4 q[2]; };

__device__ __forceinline__ unsigned short f2bf(float f) {
  unsigned int u = __builtin_bit_cast(unsigned int, f);
  u += 0x7FFFu + ((u >> 16) & 1u);
  return (unsigned short)(u >> 16);
}
__device__ __forceinline__ float bf2f(unsigned short s) {
  unsigned int u = ((unsigned int)s) << 16;
  return __builtin_bit_cast(float, u);
}
__device__ __forceinline__ float sigf(float x) {
  return 1.0f / (1.0f + __expf(-x));
}

// A fragment from a 16-row LDS panel (stride LDA u16), K-block at k0.
// Lanes 0-15: M=l, K k0+[0..7] & k0+[16..23]; lanes 16-31: M=l-16, +8.
__device__ __forceinline__ v16bf fragA(const unsigned short* lds, int k0, int lane) {
  int m  = lane & 15;
  int kh = (lane >> 4) * 8;
  const unsigned short* p = lds + m * LDA + k0 + kh;
  FragAB f;
  f.q[0] = *(const uint4*)(p);
  f.q[1] = *(const uint4*)(p + 16);
  return f.b;
}

// B fragment for out-column ncol: B[k][n] = W[ncol][k0+k], W row-major ld=256.
// Lanes 0-15: K 0-7 & 16-23 ; lanes 16-31: K 8-15 & 24-31 (within the 32-block).
__device__ __forceinline__ v16bf fragB(const unsigned short* W, int ncol, int k0, int lane) {
  const unsigned short* wr = W + (size_t)ncol * 256 + k0 + ((lane >> 4) * 8);
  FragAB f;
  f.q[0] = ((const uint4*)wr)[0];
  f.q[1] = ((const uint4*)wr)[2];
  return f.b;
}

__global__ void k_f32_to_bf16(const float* __restrict__ src,
                              unsigned short* __restrict__ dst, long n) {
  long i = (long)blockIdx.x * blockDim.x + threadIdx.x;
  long st = (long)gridDim.x * blockDim.x;
  for (; i < n; i += st) dst[i] = f2bf(src[i]);
}
__global__ void k_zero_f32(float* __restrict__ p, long n) {
  long i = (long)blockIdx.x * blockDim.x + threadIdx.x;
  long st = (long)gridDim.x * blockDim.x;
  for (; i < n; i += st) p[i] = 0.0f;
}
__global__ void k_zero_u16(unsigned short* __restrict__ p, long n) {
  long i = (long)blockIdx.x * blockDim.x + threadIdx.x;
  long st = (long)gridDim.x * blockDim.x;
  for (; i < n; i += st) p[i] = 0;
}

// Per-child forget-gate GEMM + segment-sum of (h_ch, f*c_ch) into parent slots.
// Block: 512 threads (16 waves), 32 child rows, 256 H-cols (16 per wave).
__global__ __launch_bounds__(512)
void k_child_reduce(const unsigned short* __restrict__ hbf,
                    const float* __restrict__ cbuf,
                    const int* __restrict__ par,
                    const unsigned short* __restrict__ ufw,
                    const float* __restrict__ ufb,
                    float* __restrict__ htild,
                    float* __restrict__ cagg,
                    int nd /* parents per tree */) {
  const int nc = 2 * nd;          // children per tree (level d+1)
  const int cstart = nc - 1;      // STARTS[d+1]
  const int pstart = nd - 1;      // STARTS[d]
  __shared__ alignas(16) unsigned short lA[32 * LDA];
  __shared__ int lNid[32];
  int tid = threadIdx.x, wave = tid >> 5, lane = tid & 31;
  int rowbase = blockIdx.x * 32;
  if (tid < 32) {
    int r = rowbase + tid;
    int g = r / nc, idx = r - g * nc;
    lNid[tid] = g * NPG_ + cstart + idx;
  }
  __syncthreads();
  // Stage the entire 32x256 bf16 h_ch panel once (uint4 = 8 u16 per chunk).
  for (int i = tid; i < 32 * 32; i += 512) {
    int row = i >> 5, c8 = (i & 31) << 3;
    *(uint4*)(lA + row * LDA + c8) =
        *(const uint4*)(hbf + (size_t)lNid[row] * HS + c8);
  }
  __syncthreads();
  v8f acc0 = {}, acc1 = {};
  int ncol = wave * 16 + (lane & 15);
#pragma unroll
  for (int k0 = 0; k0 < HS; k0 += 32) {
    v16bf a0 = fragA(lA, k0, lane);
    v16bf a1 = fragA(lA + 16 * LDA, k0, lane);
    v16bf b  = fragB(ufw, ncol, k0, lane);
    acc0 = __builtin_amdgcn_wmma_f32_16x16x32_bf16(false, a0, false, b, (short)0, acc0, false, false);
    acc1 = __builtin_amdgcn_wmma_f32_16x16x32_bf16(false, a1, false, b, (short)0, acc1, false, false);
  }
  float bias = ufb[ncol];
#pragma unroll
  for (int v = 0; v < 8; ++v) {
    int m = v + ((lane >> 4) << 3);
#pragma unroll
    for (int sub = 0; sub < 2; ++sub) {
      int mm = m + sub * 16;
      int nid = lNid[mm];
      float fg = sigf((sub ? acc1[v] : acc0[v]) + bias);
      float cch = cbuf[(size_t)nid * HS + ncol];
      int p = par[nid];
      int seg = (p / NPG_) * nd + (p % NPG_ - pstart);
      atomicAdd(&cagg[(size_t)seg * HS + ncol], fg * cch);
      atomicAdd(&htild[(size_t)seg * HS + ncol], bf2f(lA[mm * LDA + ncol]));
    }
  }
}

// Fused node update: iou = emb[x] @ W_iou^T (+ htild @ U_iou^T) + b; gates;
// writes h (bf16) and c (f32). Block: 512 threads, 32 node rows, 768 cols
// (wave w owns i/o/u columns 16w, 256+16w, 512+16w for both 16-row subtiles).
__global__ __launch_bounds__(512)
void k_node_update(const int* __restrict__ x,
                   const unsigned short* __restrict__ embbf,
                   const unsigned short* __restrict__ wiou,
                   const unsigned short* __restrict__ uiou,
                   const float* __restrict__ biou,
                   const unsigned short* __restrict__ htildbf,
                   const float* __restrict__ cagg,
                   unsigned short* __restrict__ hbf,
                   float* __restrict__ cbuf,
                   int nd, int hasChildren) {
  const int start = nd - 1;       // STARTS[d]
  __shared__ alignas(16) unsigned short lA[32 * LDA];
  __shared__ int lNid[32];
  __shared__ int lTok[32];
  int tid = threadIdx.x, wave = tid >> 5, lane = tid & 31;
  int rowbase = blockIdx.x * 32;
  if (tid < 32) {
    int r = rowbase + tid;
    int g = r / nd, idx = r - g * nd;
    int nid = g * NPG_ + start + idx;
    lNid[tid] = nid;
    lTok[tid] = x[nid];
  }
  __syncthreads();
  // Stage the 32x256 bf16 embedding panel once.
  for (int i = tid; i < 32 * 32; i += 512) {
    int row = i >> 5, c8 = (i & 31) << 3;
    *(uint4*)(lA + row * LDA + c8) =
        *(const uint4*)(embbf + (size_t)lTok[row] * ES + c8);
  }
  __syncthreads();
  v8f ai0 = {}, ao0 = {}, au0 = {}, ai1 = {}, ao1 = {}, au1 = {};
  int nl = lane & 15;
  int cb = wave * 16;
#pragma unroll
  for (int k0 = 0; k0 < ES; k0 += 32) {
    v16bf a0 = fragA(lA, k0, lane);
    v16bf a1 = fragA(lA + 16 * LDA, k0, lane);
    v16bf bi = fragB(wiou, cb + nl,       k0, lane);
    v16bf bo = fragB(wiou, cb + nl + 256, k0, lane);
    v16bf bu = fragB(wiou, cb + nl + 512, k0, lane);
    ai0 = __builtin_amdgcn_wmma_f32_16x16x32_bf16(false, a0, false, bi, (short)0, ai0, false, false);
    ao0 = __builtin_amdgcn_wmma_f32_16x16x32_bf16(false, a0, false, bo, (short)0, ao0, false, false);
    au0 = __builtin_amdgcn_wmma_f32_16x16x32_bf16(false, a0, false, bu, (short)0, au0, false, false);
    ai1 = __builtin_amdgcn_wmma_f32_16x16x32_bf16(false, a1, false, bi, (short)0, ai1, false, false);
    ao1 = __builtin_amdgcn_wmma_f32_16x16x32_bf16(false, a1, false, bo, (short)0, ao1, false, false);
    au1 = __builtin_amdgcn_wmma_f32_16x16x32_bf16(false, a1, false, bu, (short)0, au1, false, false);
  }
  if (hasChildren) {
    __syncthreads();
    // Restage with the 32x256 bf16 h_tild panel (level-slot rows).
    for (int i = tid; i < 32 * 32; i += 512) {
      int row = i >> 5, c8 = (i & 31) << 3;
      *(uint4*)(lA + row * LDA + c8) =
          *(const uint4*)(htildbf + (size_t)(rowbase + row) * HS + c8);
    }
    __syncthreads();
#pragma unroll
    for (int k0 = 0; k0 < HS; k0 += 32) {
      v16bf a0 = fragA(lA, k0, lane);
      v16bf a1 = fragA(lA + 16 * LDA, k0, lane);
      v16bf bi = fragB(uiou, cb + nl,       k0, lane);
      v16bf bo = fragB(uiou, cb + nl + 256, k0, lane);
      v16bf bu = fragB(uiou, cb + nl + 512, k0, lane);
      ai0 = __builtin_amdgcn_wmma_f32_16x16x32_bf16(false, a0, false, bi, (short)0, ai0, false, false);
      ao0 = __builtin_amdgcn_wmma_f32_16x16x32_bf16(false, a0, false, bo, (short)0, ao0, false, false);
      au0 = __builtin_amdgcn_wmma_f32_16x16x32_bf16(false, a0, false, bu, (short)0, au0, false, false);
      ai1 = __builtin_amdgcn_wmma_f32_16x16x32_bf16(false, a1, false, bi, (short)0, ai1, false, false);
      ao1 = __builtin_amdgcn_wmma_f32_16x16x32_bf16(false, a1, false, bo, (short)0, ao1, false, false);
      au1 = __builtin_amdgcn_wmma_f32_16x16x32_bf16(false, a1, false, bu, (short)0, au1, false, false);
    }
  }
  int n = cb + nl;
  float bi_ = biou[n], bo_ = biou[256 + n], bu_ = biou[512 + n];
#pragma unroll
  for (int v = 0; v < 8; ++v) {
    int m = v + ((lane >> 4) << 3);
#pragma unroll
    for (int sub = 0; sub < 2; ++sub) {
      int mm = m + sub * 16;
      int r = rowbase + mm;
      int nid = lNid[mm];
      float iv = (sub ? ai1[v] : ai0[v]) + bi_;
      float ov = (sub ? ao1[v] : ao0[v]) + bo_;
      float uv = (sub ? au1[v] : au0[v]) + bu_;
      float cn = sigf(iv) * tanhf(uv);
      if (hasChildren) cn += cagg[(size_t)r * HS + n];
      float hn = sigf(ov) * tanhf(cn);
      cbuf[(size_t)nid * HS + n] = cn;
      hbf[(size_t)nid * HS + n] = f2bf(hn);
    }
  }
}

// Root classifier: [32 roots, 256] @ lin_w^T [256,104] + lin_b.
// One block, 14 active waves = 2 row-tiles x 7 col-tiles (col mask at 104).
__global__ __launch_bounds__(512)
void k_out(const unsigned short* __restrict__ hbf,
           const unsigned short* __restrict__ linw,
           const float* __restrict__ linb,
           float* __restrict__ out) {
  int tid = threadIdx.x, wave = tid >> 5, lane = tid & 31;
  if (wave >= 14) return;
  int rt = wave / 7, ct = wave % 7;
  int nl = lane & 15;
  int ncol = ct * 16 + nl;
  v8f acc = {};
#pragma unroll
  for (int k0 = 0; k0 < HS; k0 += 32) {
    FragAB fa;
    int m = lane & 15;
    int g = rt * 16 + m;
    int kh = (lane >> 4) * 8;
    const unsigned short* hr = hbf + (size_t)g * NPG_ * HS + k0 + kh;
    fa.q[0] = *(const uint4*)(hr);
    fa.q[1] = *(const uint4*)(hr + 16);
    FragAB fb;
    if (ncol < NCLS) {
      const unsigned short* wr = linw + (size_t)ncol * HS + k0 + ((lane >> 4) * 8);
      fb.q[0] = ((const uint4*)wr)[0];
      fb.q[1] = ((const uint4*)wr)[2];
    } else {
      fb.q[0] = make_uint4(0, 0, 0, 0);
      fb.q[1] = make_uint4(0, 0, 0, 0);
    }
    acc = __builtin_amdgcn_wmma_f32_16x16x32_bf16(false, fa.b, false, fb.b,
                                                  (short)0, acc, false, false);
  }
  if (ncol < NCLS) {
#pragma unroll
    for (int v = 0; v < 8; ++v) {
      int m = v + ((lane >> 4) << 3);
      int g = rt * 16 + m;
      out[g * NCLS + ncol] = acc[v] + linb[ncol];
    }
  }
}

extern "C" void kernel_launch(void* const* d_in, const int* in_sizes, int n_in,
                              void* d_out, int out_size, void* d_ws, size_t ws_size,
                              hipStream_t stream) {
  (void)in_sizes; (void)n_in; (void)out_size; (void)ws_size;
  const int*   x    = (const int*)d_in[0];
  const int*   par  = (const int*)d_in[1];
  const float* emb  = (const float*)d_in[2];
  const float* Wiou = (const float*)d_in[3];
  const float* Uiou = (const float*)d_in[4];
  const float* biou = (const float*)d_in[5];
  const float* Ufw  = (const float*)d_in[6];
  const float* Ufb  = (const float*)d_in[7];
  const float* linw = (const float*)d_in[8];
  const float* linb = (const float*)d_in[9];
  float* out = (float*)d_out;

  char* base = (char*)d_ws;
  size_t off = 0;
  auto carve = [&](size_t bytes) -> void* {
    void* p = base + off;
    off += (bytes + 255) & ~(size_t)255;
    return p;
  };
  unsigned short* hbf     = (unsigned short*)carve((size_t)NTOT * HS * 2);
  float*          cbuf    = (float*)carve((size_t)NTOT * HS * 4);
  float*          htild   = (float*)carve((size_t)32768 * HS * 4);
  float*          cagg    = (float*)carve((size_t)32768 * HS * 4);
  unsigned short* htildbf = (unsigned short*)carve((size_t)32768 * HS * 2);
  unsigned short* embbf   = (unsigned short*)carve((size_t)VOCAB_ * ES * 2);
  unsigned short* wioubf  = (unsigned short*)carve((size_t)H3 * ES * 2);
  unsigned short* uioubf  = (unsigned short*)carve((size_t)H3 * HS * 2);
  unsigned short* ufwbf   = (unsigned short*)carve((size_t)HS * HS * 2);
  unsigned short* linwbf  = (unsigned short*)carve((size_t)NCLS * HS * 2);

  auto conv = [&](const float* s, unsigned short* dptr, long n) {
    int blocks = (int)((n + 255) / 256);
    if (blocks > 4096) blocks = 4096;
    k_f32_to_bf16<<<blocks, 256, 0, stream>>>(s, dptr, n);
  };
  conv(emb, embbf, (long)VOCAB_ * ES);
  k_zero_u16<<<1, 256, 0, stream>>>(embbf + (size_t)PADTOK * ES, (long)ES);
  conv(Wiou, wioubf, (long)H3 * ES);
  conv(Uiou, uioubf, (long)H3 * HS);
  conv(Ufw, ufwbf, (long)HS * HS);
  conv(linw, linwbf, (long)NCLS * HS);

  // Leaves (level 11): no children, c_agg = 0.
  {
    int nd = 1 << 11;
    int rows = TREES * nd;
    k_node_update<<<rows / 32, 512, 0, stream>>>(x, embbf, wioubf, uioubf, biou,
                                                 htildbf, cagg, hbf, cbuf, nd, 0);
  }
  for (int d = 10; d >= 0; --d) {
    int nd = 1 << d;
    long pElems = (long)TREES * nd * HS;
    {
      int blocks = (int)((pElems + 255) / 256);
      if (blocks > 4096) blocks = 4096;
      k_zero_f32<<<blocks, 256, 0, stream>>>(htild, pElems);
      k_zero_f32<<<blocks, 256, 0, stream>>>(cagg, pElems);
    }
    int crows = TREES * 2 * nd;
    k_child_reduce<<<crows / 32, 512, 0, stream>>>(hbf, cbuf, par, ufwbf, Ufb,
                                                   htild, cagg, nd);
    conv(htild, htildbf, pElems);
    int rows = TREES * nd;
    k_node_update<<<rows / 32, 512, 0, stream>>>(x, embbf, wioubf, uioubf, biou,
                                                 htildbf, cagg, hbf, cbuf, nd, 1);
  }
  k_out<<<1, 512, 0, stream>>>(hbf, linwbf, linb, out);
}
